// Gj_40716289966841
// MI455X (gfx1250) — matrix operand order repaired and checked by reading, not compiled, not run on hardware
//
#include <hip/hip_runtime.h>

// ---------------------------------------------------------------------------
// Per-species expert Linear (MoE-style) for MI455X / gfx1250.
//   out[i] = rho[i].flatten() @ W[sym[i]].T + b[sym[i]]
//
//   1) counting-sort atom indices by species (segments padded to 64, pad=-1)
//   2) pre-split W (f32) -> Whi/Wlo (bf16) once per launch in workspace
//   3) GEMM: workgroup = 64 gathered rows x 512 cols of ONE expert
//      - X slab split to bf16 hi/lo *during* LDS staging (converted once,
//        consumed by all 8 waves via ds_load_b128 fragments)
//      - f32 emulated as bf16 hi/lo: 3x v_wmma_f32_16x16x32_bf16 per tile
//      - products ordered outer / row-tiles inner -> independent WMMA chains
// ---------------------------------------------------------------------------

#define DF 512
#define XS 40  // LDS row stride in bf16: 80B = 20 banks -> conflict-free b128

typedef __attribute__((ext_vector_type(16))) __bf16 v16bf;
typedef __attribute__((ext_vector_type(8)))  __bf16 v8bf;
typedef __attribute__((ext_vector_type(8)))  float  v8f;

union V16 {
  v16bf v;
  v8bf  h[2];
};

__device__ __forceinline__ void split_bf16(float x, __bf16& hi, __bf16& lo) {
  hi = (__bf16)x;                // RTNE to bf16
  lo = (__bf16)(x - (float)hi);  // residual (restores ~16 mantissa bits)
}

// ---- sort pipeline --------------------------------------------------------

__global__ void k_init(int* __restrict__ perm, unsigned* __restrict__ cnt, int permN) {
  int i = blockIdx.x * 256 + threadIdx.x;
  if (i < permN) perm[i] = -1;
  if (i < 4) cnt[i] = 0u;
}

__global__ void k_hist(const int* __restrict__ sym, unsigned* __restrict__ cnt, int N) {
  __shared__ unsigned lc[4];
  if (threadIdx.x < 4) lc[threadIdx.x] = 0u;
  __syncthreads();
  int i = blockIdx.x * 256 + threadIdx.x;
  if (i < N) atomicAdd(&lc[sym[i] & 3], 1u);
  __syncthreads();
  if (threadIdx.x < 4 && lc[threadIdx.x]) atomicAdd(&cnt[threadIdx.x], lc[threadIdx.x]);
}

__global__ void k_prefix(const unsigned* __restrict__ cnt,
                         unsigned* __restrict__ cursor,
                         unsigned* __restrict__ pstart) {
  if (threadIdx.x == 0) {
    unsigned off = 0;
    for (int s = 0; s < 4; ++s) {
      pstart[s] = off;
      off = (off + cnt[s] + 63u) & ~63u;  // pad each segment to 64 rows
      cursor[s] = 0u;
    }
  }
}

__global__ void k_scatter(const int* __restrict__ sym, unsigned* __restrict__ cursor,
                          const unsigned* __restrict__ pstart, int* __restrict__ perm, int N) {
  __shared__ unsigned lc[4];
  __shared__ unsigned gb[4];
  if (threadIdx.x < 4) lc[threadIdx.x] = 0u;
  __syncthreads();
  int i = blockIdx.x * 256 + threadIdx.x;
  int s = 0; unsigned lpos = 0;
  if (i < N) { s = sym[i] & 3; lpos = atomicAdd(&lc[s], 1u); }
  __syncthreads();
  if (threadIdx.x < 4)
    gb[threadIdx.x] = lc[threadIdx.x] ? atomicAdd(&cursor[threadIdx.x], lc[threadIdx.x]) : 0u;
  __syncthreads();
  if (i < N) perm[pstart[s] + gb[s] + lpos] = i;
}

// ---- W -> bf16 hi/lo pre-split (once per launch; W is tiny vs the stream) --

__global__ void k_wsplit(const float* __restrict__ Wm, __bf16* __restrict__ whi,
                         __bf16* __restrict__ wlo, int total) {
  int i = blockIdx.x * 256 + threadIdx.x;
  if (i < total) {
    float w = Wm[i];
    __bf16 hi, lo;
    split_bf16(w, hi, lo);
    whi[i] = hi;
    wlo[i] = lo;
  }
}

// ---- GEMM kernel ----------------------------------------------------------

__global__ void __launch_bounds__(256)
k_moe_gemm(const float* __restrict__ rho, const __bf16* __restrict__ whi,
           const __bf16* __restrict__ wlo, const float* __restrict__ bv,
           const int* __restrict__ perm, const unsigned* __restrict__ pstart,
           float* __restrict__ out) {
  __shared__ alignas(16) __bf16 xsh[64 * XS];  // X slab hi plane
  __shared__ alignas(16) __bf16 xsl[64 * XS];  // X slab lo plane
  __shared__ int prow[64];

  const unsigned base = blockIdx.x * 64u;
  int s = 0;
  if (base >= pstart[1]) s = 1;
  if (base >= pstart[2]) s = 2;
  if (base >= pstart[3]) s = 3;

  if (threadIdx.x < 64) prow[threadIdx.x] = perm[base + threadIdx.x];
  __syncthreads();

  const int lane = threadIdx.x & 31;
  const int wv   = threadIdx.x >> 5;    // wave 0..7 -> columns [wv*64, wv*64+64)
  const int l16  = lane & 15;
  const int lhi  = lane >> 4;           // half-wave select
  const int kb   = lhi * 8;             // A fragment K base per ISA layout

  const int srow  = threadIdx.x >> 2;        // staging: row 0..63
  const int skoff = (threadIdx.x & 3) * 8;   // staging: k offset 0/8/16/24
  const int sper  = prow[srow];

  const __bf16* __restrict__ Wh = whi + (size_t)s * DF * DF;
  const __bf16* __restrict__ Wl = wlo + (size_t)s * DF * DF;

  v8f acc[4][4];
  {
    v8f z = {};
#pragma unroll
    for (int r = 0; r < 4; ++r)
#pragma unroll
      for (int c = 0; c < 4; ++c) acc[r][c] = z;
  }

  for (int kt = 0; kt < 16; ++kt) {
    const int kbase = kt * 32;

    // ---- stage 64x32 slab: gather f32, split to bf16 hi/lo planes (once) ----
    {
      v8bf hv, lv;
      if (sper >= 0) {
        const float4* p =
            reinterpret_cast<const float4*>(rho + (size_t)sper * DF + kbase + skoff);
        float4 a = p[0], b2 = p[1];
        float v[8] = {a.x, a.y, a.z, a.w, b2.x, b2.y, b2.z, b2.w};
#pragma unroll
        for (int j = 0; j < 8; ++j) {
          __bf16 hi, lo;
          split_bf16(v[j], hi, lo);
          hv[j] = hi;
          lv[j] = lo;
        }
      } else {
        hv = (v8bf)(__bf16)0.0f;
        lv = (v8bf)(__bf16)0.0f;
      }
      *reinterpret_cast<v8bf*>(&xsh[srow * XS + skoff]) = hv;
      *reinterpret_cast<v8bf*>(&xsl[srow * XS + skoff]) = lv;
    }
    __syncthreads();

    // ---- A fragments via ds_load_b128 (ISA 16-bit A layout: lanes0-15 hold
    //      K{kb..kb+7, kb+16..kb+23} with kb=0, lanes16-31 with kb=8) ----
    V16 ah[4], al[4];
#pragma unroll
    for (int r = 0; r < 4; ++r) {
      const int ro = (r * 16 + l16) * XS + kb;
      ah[r].h[0] = *reinterpret_cast<const v8bf*>(&xsh[ro]);
      ah[r].h[1] = *reinterpret_cast<const v8bf*>(&xsh[ro + 16]);
      al[r].h[0] = *reinterpret_cast<const v8bf*>(&xsl[ro]);
      al[r].h[1] = *reinterpret_cast<const v8bf*>(&xsl[ro + 16]);
    }

    // ---- column tiles: B = W[s]^T fragment (32x16 bf16: lane=N, lanes0-15
    //      K{0..7,16..23}, lanes16-31 K{8..15,24..31}) loaded directly as bf16 ----
#pragma unroll
    for (int c = 0; c < 4; ++c) {
      const int n = wv * 64 + c * 16 + l16;
      const size_t wo = (size_t)n * DF + kbase + lhi * 8;
      V16 bh, bl;
      bh.h[0] = *reinterpret_cast<const v8bf*>(Wh + wo);
      bh.h[1] = *reinterpret_cast<const v8bf*>(Wh + wo + 16);
      bl.h[0] = *reinterpret_cast<const v8bf*>(Wl + wo);
      bl.h[1] = *reinterpret_cast<const v8bf*>(Wl + wo + 16);

      // product-major order: consecutive WMMAs target different accumulators
#pragma unroll
      for (int r = 0; r < 4; ++r)
        acc[r][c] = __builtin_amdgcn_wmma_f32_16x16x32_bf16(
            false, ah[r].v, false, bh.v, (short)0, acc[r][c], false, false);
#pragma unroll
      for (int r = 0; r < 4; ++r)
        acc[r][c] = __builtin_amdgcn_wmma_f32_16x16x32_bf16(
            false, ah[r].v, false, bl.v, (short)0, acc[r][c], false, false);
#pragma unroll
      for (int r = 0; r < 4; ++r)
        acc[r][c] = __builtin_amdgcn_wmma_f32_16x16x32_bf16(
            false, al[r].v, false, bh.v, (short)0, acc[r][c], false, false);
    }
    __syncthreads();
  }

  // ---- bias add + scattered store (C/D layout: VGPR j -> M=j / j+8) ----
  const float* bs = bv + (size_t)s * DF;
#pragma unroll
  for (int c = 0; c < 4; ++c) {
    const int n = wv * 64 + c * 16 + l16;
    const float bias = bs[n];
#pragma unroll
    for (int r = 0; r < 4; ++r) {
#pragma unroll
      for (int j = 0; j < 8; ++j) {
        const int row = r * 16 + j + lhi * 8;
        const int pr  = prow[row];
        if (pr >= 0) out[(size_t)pr * DF + n] = acc[r][c][j] + bias;
      }
    }
  }
}

// ---- launcher -------------------------------------------------------------

extern "C" void kernel_launch(void* const* d_in, const int* in_sizes, int n_in,
                              void* d_out, int out_size, void* d_ws, size_t ws_size,
                              hipStream_t stream) {
  const float* rho = (const float*)d_in[0];
  const float* W   = (const float*)d_in[1];
  const float* b   = (const float*)d_in[2];
  const int*   sym = (const int*)d_in[3];
  float* out = (float*)d_out;

  const int N  = in_sizes[0] / DF;
  const int NW = in_sizes[1];  // total W elements (4*512*512)

  const int ngroups = (N + 315) / 64;  // covers sum of 64-aligned segments
  const int permN   = ngroups * 64;    // multiple of 64 -> keeps alignment

  unsigned* cnt    = (unsigned*)d_ws;       // [0..3]  counts
  unsigned* cursor = cnt + 4;               // [4..7]  scatter cursors
  unsigned* pstart = cnt + 8;               // [8..11] padded segment starts
  int*    perm = (int*)d_ws + 64;           // padded permutation (256B offset)
  __bf16* whi  = (__bf16*)(perm + permN);   // W hi plane (bf16)
  __bf16* wlo  = whi + NW;                  // W lo plane (bf16)

  k_init<<<(permN + 255) / 256, 256, 0, stream>>>(perm, cnt, permN);
  k_hist<<<(N + 255) / 256, 256, 0, stream>>>(sym, cnt, N);
  k_prefix<<<1, 32, 0, stream>>>(cnt, cursor, pstart);
  k_scatter<<<(N + 255) / 256, 256, 0, stream>>>(sym, cursor, pstart, perm, N);
  k_wsplit<<<(NW + 255) / 256, 256, 0, stream>>>(W, whi, wlo, NW);
  k_moe_gemm<<<ngroups, 256, 0, stream>>>(rho, whi, wlo, b, perm, pstart, out);
}
